// BroadBINLayer_25331717111832
// MI455X (gfx1250) — matrix operand order repaired and checked by reading, not compiled, not run on hardware
//
#include <hip/hip_runtime.h>
#include <hip/hip_bf16.h>

typedef __attribute__((ext_vector_type(16))) _Float16 v16h;
typedef __attribute__((ext_vector_type(8)))  float    v8f;

// Explicit global-address-space pointers so B-tile loads lower to
// global_load_b128 (LOADcnt only) instead of flat_load_b128 (LOADcnt+DScnt).
typedef const __attribute__((address_space(1))) _Float16* gptr_f16;
typedef const __attribute__((address_space(1))) v16h*     gptr_v16h;

#define NUM_GROUPS 100
#define GROUP_SIZE 100
#define TOTAL_ROWS 10000
#define NUM_CLASSES 1000
#define BATCH 4096
#define CLS_PAD 1024          // 64 tiles of 16 classes -> branch-free tile loop
#define GRP_PAD 128           // group rows padded 100 -> 128 (4 K-chunks of 32)
#define ROW_PAD (NUM_GROUPS * GRP_PAD)   // 12800 padded rows
#define NUM_TILES 64          // 8 waves x 8 tiles, no remainder

// ---------------------------------------------------------------------------
// Kernel 1: per-(group,class) online softmax stats over the 100 group rows.
// m[g][c] = max_r W[g*100+r][c] ; l[g][c] = sum_r exp(W - m)
// ---------------------------------------------------------------------------
__global__ __launch_bounds__(256) void group_stats(const float* __restrict__ Wl,
                                                   float* __restrict__ m,
                                                   float* __restrict__ l) {
    int c = blockIdx.x * 256 + threadIdx.x;   // class
    int g = blockIdx.y;                       // group
    if (c >= NUM_CLASSES) return;
    const float* p = Wl + (size_t)g * GROUP_SIZE * NUM_CLASSES + c;
    float mx = -3.402823466e38f, s = 0.f;
    for (int r = 0; r < GROUP_SIZE; ++r) {
        float w  = p[(size_t)r * NUM_CLASSES];
        float nm = fmaxf(mx, w);
        s = s * __expf(mx - nm) + __expf(w - nm);
        mx = nm;
    }
    m[g * CLS_PAD + c] = mx;
    l[g * CLS_PAD + c] = s;
}

// ---------------------------------------------------------------------------
// Kernel 2: build transposed f16 table Wt[c][padded_row] = log(theta + eps).
// Padding (local row >= 100 or class >= 1000) is hard zero, so padded K-lanes
// and padded classes contribute nothing to the WMMA accumulation.
// ---------------------------------------------------------------------------
__global__ __launch_bounds__(256) void build_wt(const float* __restrict__ Wl,
                                                const float* __restrict__ m,
                                                const float* __restrict__ l,
                                                _Float16* __restrict__ Wt) {
    size_t idx = (size_t)blockIdx.x * 256 + threadIdx.x;
    if (idx >= (size_t)CLS_PAD * ROW_PAD) return;
    int c = (int)(idx / ROW_PAD);
    int r = (int)(idx % ROW_PAD);
    int g = r >> 7;            // /128
    int j = r & 127;           // local padded row
    _Float16 v = (_Float16)0.f;
    if (c < NUM_CLASSES && j < GROUP_SIZE) {
        float w     = Wl[(size_t)(g * GROUP_SIZE + j) * NUM_CLASSES + c];
        float theta = __expf(w - m[g * CLS_PAD + c]) / l[g * CLS_PAD + c];
        v = (_Float16)__logf(theta + 1e-12f);
    }
    Wt[idx] = v;
}

// ---------------------------------------------------------------------------
// Kernel 3: recover local row index per (sample, group) from the one-hot.
// One wave per pair; the single matching lane does the store.
// ---------------------------------------------------------------------------
__global__ __launch_bounds__(256) void extract_idx(const float* __restrict__ x,
                                                   int* __restrict__ idxArr) {
    int wave = threadIdx.x >> 5;
    int lane = threadIdx.x & 31;
    int pair = blockIdx.x * 8 + wave;             // (b,g) pair
    if (pair >= BATCH * NUM_GROUPS) return;
    int b = pair / NUM_GROUPS;
    int g = pair % NUM_GROUPS;
    const float* p = x + (size_t)b * TOTAL_ROWS + g * GROUP_SIZE;
    for (int j = lane; j < GROUP_SIZE; j += 32)
        if (p[j] > 0.5f) idxArr[pair] = j;        // exactly one lane fires
}

// ---------------------------------------------------------------------------
// Kernel 4: WMMA gather-GEMM + bias + class softmax.
// Grid: 256 WGs (16 samples each). 8 waves; wave w owns class tiles
// t = w + 8*j for j = 0..7 (all valid: 64 tiles). Branch-free inner loop:
// build one-hot A (f16 16x32), 8 unguarded B loads + 8 WMMAs per K-chunk.
// EXEC is all-ones throughout (no divergent control flow around WMMA).
// ---------------------------------------------------------------------------
__global__ __launch_bounds__(256) void logits_softmax(const _Float16* __restrict__ Wt,
                                                      const int* __restrict__ idxArr,
                                                      const float* __restrict__ bias,
                                                      float* __restrict__ out) {
    __shared__ int   idxLds[16 * NUM_GROUPS];
    __shared__ float redmax[8][16];
    __shared__ float redsum[8][16];

    const int tid  = threadIdx.x;
    const int wv   = __builtin_amdgcn_readfirstlane(tid >> 5);  // SGPR wave id
    const int lane = tid & 31;
    const int half = lane >> 4;      // 0: lanes 0-15, 1: lanes 16-31
    const int ln   = lane & 15;
    const int b0   = blockIdx.x * 16;

    // Stage the 16 samples' 100 indices in LDS.
    for (int i = tid; i < 16 * NUM_GROUPS; i += 256) {
        int s = i / NUM_GROUPS, g = i % NUM_GROUPS;
        idxLds[i] = idxArr[(size_t)(b0 + s) * NUM_GROUPS + g];
    }
    __syncthreads();

    v8f acc[8];
    #pragma unroll
    for (int j = 0; j < 8; ++j)
        #pragma unroll
        for (int i = 0; i < 8; ++i) acc[j][i] = 0.f;

    // Global-AS base + per-tile induction pointers (advanced once per group);
    // chunk offset folds into the load's immediate offset field.
    gptr_f16 Wg = (gptr_f16)Wt;
    gptr_f16 p[8];
    #pragma unroll
    for (int j = 0; j < 8; ++j) {
        const int t = wv + 8 * j;    // always < 64
        p[j] = Wg + (size_t)(t * 16 + ln) * ROW_PAD + half * 16;
    }

    const int kb = half * 16;  // lane-half K base for the A layout

    for (int g = 0; g < NUM_GROUPS; ++g) {
        // A-matrix row M = ln; fold the lane-dependent K base in once.
        const int lk = idxLds[ln * NUM_GROUPS + g] - kb;
        #pragma unroll
        for (int ch = 0; ch < 4; ++ch) {
            // One-hot A, contiguous K layout matching B:
            // lanes 0-15: halves 0..15 -> K=0..15
            // lanes16-31: halves 0..15 -> K=16..31
            const int d = lk - ch * 32;
            v16h A;
            #pragma unroll
            for (int h = 0; h < 16; ++h) {
                const int hK = h;   // inline-constant target
                A[h] = (d == hK) ? (_Float16)1.f : (_Float16)0.f;
            }
            #pragma unroll
            for (int j = 0; j < 8; ++j) {
                // B 32x16: lane holds column N=ln, 16 contiguous K halves
                v16h B = *(gptr_v16h)(p[j] + ch * 32);
                acc[j] = __builtin_amdgcn_wmma_f32_16x16x32_f16(
                    false, A, false, B, (short)0, acc[j], false, false);
            }
        }
        #pragma unroll
        for (int j = 0; j < 8; ++j) p[j] += GRP_PAD;     // next group slab
    }

    // Bias add (class c = t*16 + ln, one class per lane per tile).
    #pragma unroll
    for (int j = 0; j < 8; ++j) {
        const int c = (wv + 8 * j) * 16 + ln;
        if (c < NUM_CLASSES) {
            const float bv = bias[c];
            #pragma unroll
            for (int i = 0; i < 8; ++i) acc[j][i] += bv;
        }
    }

    // ---- softmax over classes (per sample s = half*8 + i) ----
    float pm[8];
    #pragma unroll
    for (int i = 0; i < 8; ++i) pm[i] = -3.402823466e38f;
    #pragma unroll
    for (int j = 0; j < 8; ++j) {
        const int c = (wv + 8 * j) * 16 + ln;
        if (c < NUM_CLASSES)
            #pragma unroll
            for (int i = 0; i < 8; ++i) pm[i] = fmaxf(pm[i], acc[j][i]);
    }
    for (int off = 1; off < 16; off <<= 1)
        #pragma unroll
        for (int i = 0; i < 8; ++i) pm[i] = fmaxf(pm[i], __shfl_xor(pm[i], off, 32));
    if (ln == 0)
        #pragma unroll
        for (int i = 0; i < 8; ++i) redmax[wv][half * 8 + i] = pm[i];
    __syncthreads();
    float gm[8];
    #pragma unroll
    for (int i = 0; i < 8; ++i) {
        float v = redmax[0][half * 8 + i];
        for (int w = 1; w < 8; ++w) v = fmaxf(v, redmax[w][half * 8 + i]);
        gm[i] = v;
    }

    float ps[8];
    #pragma unroll
    for (int i = 0; i < 8; ++i) ps[i] = 0.f;
    #pragma unroll
    for (int j = 0; j < 8; ++j) {
        const int c = (wv + 8 * j) * 16 + ln;
        if (c < NUM_CLASSES)
            #pragma unroll
            for (int i = 0; i < 8; ++i) ps[i] += __expf(acc[j][i] - gm[i]);
    }
    for (int off = 1; off < 16; off <<= 1)
        #pragma unroll
        for (int i = 0; i < 8; ++i) ps[i] += __shfl_xor(ps[i], off, 32);
    if (ln == 0)
        #pragma unroll
        for (int i = 0; i < 8; ++i) redsum[wv][half * 8 + i] = ps[i];
    __syncthreads();
    float gs[8];
    #pragma unroll
    for (int i = 0; i < 8; ++i) {
        float v = 0.f;
        for (int w = 0; w < 8; ++w) v += redsum[w][half * 8 + i];
        gs[i] = v;
    }

    // Write probabilities.
    #pragma unroll
    for (int j = 0; j < 8; ++j) {
        const int c = (wv + 8 * j) * 16 + ln;
        if (c < NUM_CLASSES)
            #pragma unroll
            for (int i = 0; i < 8; ++i)
                out[(size_t)(b0 + half * 8 + i) * NUM_CLASSES + c] =
                    __expf(acc[j][i] - gm[i]) / gs[i];
    }
}

// ---------------------------------------------------------------------------
extern "C" void kernel_launch(void* const* d_in, const int* in_sizes, int n_in,
                              void* d_out, int out_size, void* d_ws, size_t ws_size,
                              hipStream_t stream) {
    (void)in_sizes; (void)n_in; (void)out_size; (void)ws_size;
    const float* x    = (const float*)d_in[0];   // (4096, 10000)
    const float* Wl   = (const float*)d_in[1];   // (10000, 1000)
    const float* bias = (const float*)d_in[2];   // (1000,)
    float* out = (float*)d_out;                  // (4096, 1000)

    // Workspace layout (base is 256B-aligned):
    //   Wt  : CLS_PAD * ROW_PAD f16  = 26,214,400 B
    //   m   : 100 * CLS_PAD f32      =    409,600 B
    //   l   : 100 * CLS_PAD f32      =    409,600 B
    //   idx : BATCH * 100 i32        =  1,638,400 B   (total ~28.7 MB)
    char* ws = (char*)d_ws;
    _Float16* Wt = (_Float16*)(ws);
    float*    m  = (float*)(ws + (size_t)CLS_PAD * ROW_PAD * 2);
    float*    l  = (float*)(ws + (size_t)CLS_PAD * ROW_PAD * 2 + (size_t)NUM_GROUPS * CLS_PAD * 4);
    int*      ix = (int*)  (ws + (size_t)CLS_PAD * ROW_PAD * 2 + (size_t)NUM_GROUPS * CLS_PAD * 8);

    // 1) per-group column softmax stats
    group_stats<<<dim3((NUM_CLASSES + 255) / 256, NUM_GROUPS), 256, 0, stream>>>(Wl, m, l);

    // 2) f16 transposed log-softmax table (writes every padded element -> deterministic)
    {
        size_t n = (size_t)CLS_PAD * ROW_PAD;
        build_wt<<<(unsigned)((n + 255) / 256), 256, 0, stream>>>(Wl, m, l, Wt);
    }

    // 3) index extraction from the one-hot
    extract_idx<<<(BATCH * NUM_GROUPS + 7) / 8, 256, 0, stream>>>(x, ix);

    // 4) WMMA contraction + bias + class softmax
    logits_softmax<<<BATCH / 16, 256, 0, stream>>>(Wt, ix, bias, out);
}
